// FullModel_28063316312554
// MI455X (gfx1250) — matrix operand order repaired and checked by reading, not compiled, not run on hardware
//
#include <hip/hip_runtime.h>
#include <math.h>

// ---------------- model dims ----------------
#define E_EDGES   256
#define L_PTS     8
#define NIMG      (E_EDGES * L_PTS)   // 2048
#define N_NODES   300
#define NODE_DIM  64
#define EDGE_DIM  32
#define HIDDEN    128
#define OUT_DIM   20
#define LSTM_H    128
#define GATES     (4 * LSTM_H)        // 512

typedef __attribute__((ext_vector_type(16))) __bf16 v16bf;
typedef __attribute__((ext_vector_type(8)))  float  v8f;

__device__ __forceinline__ int imin(int a, int b) { return a < b ? a : b; }

// Convert two contiguous float4s (8 floats) into bf16 halves [base..base+7]
#define CVT8(dst, base, p0, p1)                         \
  do {                                                  \
    (dst)[(base) + 0] = (__bf16)(p0).x;                 \
    (dst)[(base) + 1] = (__bf16)(p0).y;                 \
    (dst)[(base) + 2] = (__bf16)(p0).z;                 \
    (dst)[(base) + 3] = (__bf16)(p0).w;                 \
    (dst)[(base) + 4] = (__bf16)(p1).x;                 \
    (dst)[(base) + 5] = (__bf16)(p1).y;                 \
    (dst)[(base) + 6] = (__bf16)(p1).z;                 \
    (dst)[(base) + 7] = (__bf16)(p1).w;                 \
  } while (0)

// =====================================================================
// WMMA GEMM:  C[M,N] = act( A[M,K] @ Wbf[N,K]^T + bias + bias2 )
//   A    : f32 activations (converted to bf16 in-register)
//   Wbf  : bf16 weights, pre-converted once per launch (direct loads)
// One wave per 16(M) x 64(N) output strip: A fragment loaded once per
// k-step, 4 B fragments, 4 back-to-back v_wmma_f32_16x16x32_bf16.
// All loads of a k-step are issued before any consumption so the
// scheduler can pipeline them (no per-load s_wait_loadcnt 0).
// Bounds handled by CLAMPING indices (padded rows/cols never stored;
// K always a multiple of 32).
// CDNA5 wave32 fragment layouts used here (same K order for A and B):
//   A (16x32): lane&15 = M row; halves j -> K = 16*(lane>>4) + j
//   B (32x16): lane&15 = N col; halves j -> K = 16*(lane>>4) + j
//   C/D      : col = lane&15 (+tile), VGPR r -> row = 8*(lane>>4) + r
// =====================================================================
__global__ __launch_bounds__(32)
void wmma_gemm_bf16(const float* __restrict__ A, const __bf16* __restrict__ W,
                    const float* __restrict__ bias, const float* __restrict__ bias2,
                    float* __restrict__ C,
                    int M, int N, int K, int accumulate, int do_relu)
{
  const int lane = threadIdx.x & 31;
  const int col0 = blockIdx.x * 64;     // 4 N-tiles of 16
  const int row0 = blockIdx.y * 16;

  const int kHalf = (lane >> 4) * 16;   // A-fragment K base (contiguous 16)
  const int kB    = (lane >> 4) * 16;   // B-fragment K base

  // Clamped source rows/cols (safe: padded outputs never stored)
  const float* Arow = A + (size_t)imin(row0 + (lane & 15), M - 1) * K;
  const __bf16* Wrow[4];
#pragma unroll
  for (int t = 0; t < 4; ++t)
    Wrow[t] = W + (size_t)imin(col0 + t * 16 + (lane & 15), N - 1) * K;

  const int crow = row0 + (lane >> 4) * 8;

  v8f acc[4];
  if (accumulate) {
#pragma unroll
    for (int t = 0; t < 4; ++t) {
      int ccol = col0 + t * 16 + (lane & 15);
#pragma unroll
      for (int r = 0; r < 8; ++r) {
        int rr = crow + r;
        acc[t][r] = (rr < M && ccol < N) ? C[(size_t)rr * N + ccol] : 0.0f;
      }
    }
  } else {
#pragma unroll
    for (int t = 0; t < 4; ++t)
#pragma unroll
      for (int r = 0; r < 8; ++r) acc[t][r] = 0.0f;
  }

  for (int k0 = 0; k0 < K; k0 += 32) {
    // ---- issue ALL loads first (4x b128 for A, 4x 32B for B) ----
    float4 a0 = *(const float4*)(Arow + k0 + kHalf);
    float4 a1 = *(const float4*)(Arow + k0 + kHalf + 4);
    float4 a2 = *(const float4*)(Arow + k0 + kHalf + 8);
    float4 a3 = *(const float4*)(Arow + k0 + kHalf + 12);
    v16bf b0 = *(const v16bf*)(Wrow[0] + k0 + kB);
    v16bf b1 = *(const v16bf*)(Wrow[1] + k0 + kB);
    v16bf b2 = *(const v16bf*)(Wrow[2] + k0 + kB);
    v16bf b3 = *(const v16bf*)(Wrow[3] + k0 + kB);

    // ---- convert A, then 4 independent WMMA chains ----
    v16bf a;
    CVT8(a, 0, a0, a1);
    CVT8(a, 8, a2, a3);
    acc[0] = __builtin_amdgcn_wmma_f32_16x16x32_bf16(false, a, false, b0, (short)0, acc[0], false, false);
    acc[1] = __builtin_amdgcn_wmma_f32_16x16x32_bf16(false, a, false, b1, (short)0, acc[1], false, false);
    acc[2] = __builtin_amdgcn_wmma_f32_16x16x32_bf16(false, a, false, b2, (short)0, acc[2], false, false);
    acc[3] = __builtin_amdgcn_wmma_f32_16x16x32_bf16(false, a, false, b3, (short)0, acc[3], false, false);
  }

#pragma unroll
  for (int t = 0; t < 4; ++t) {
    int ccol = col0 + t * 16 + (lane & 15);
    if (ccol >= N) continue;
    float bv = 0.0f;
    if (bias)  bv += bias[ccol];
    if (bias2) bv += bias2[ccol];
#pragma unroll
    for (int r = 0; r < 8; ++r) {
      int rr = crow + r;
      if (rr < M) {
        float v = acc[t][r] + bv;
        if (do_relu) v = fmaxf(v, 0.0f);
        C[(size_t)rr * N + ccol] = v;
      }
    }
  }
}

// One-time (per launch) f32 -> bf16 weight conversion
__global__ void f32_to_bf16(const float* __restrict__ in, __bf16* __restrict__ out, int n)
{
  int i = blockIdx.x * blockDim.x + threadIdx.x;
  if (i < n) out[i] = (__bf16)in[i];
}

// =====================================================================
// 3-D conv (3x3x3, SAME) helpers — direct VALU f32 (tiny channel counts)
// =====================================================================
__device__ __forceinline__ float conv_tap(const float* __restrict__ in,
                                          const float* __restrict__ w,
                                          float bias, int n, int CI, int D,
                                          int co, int z, int y, int x)
{
  const int D3 = D * D * D;
  float val = bias;
  for (int ci = 0; ci < CI; ++ci) {
    const float* ip = in + ((size_t)n * CI + ci) * D3;
    const float* wp = w  + ((size_t)co * CI + ci) * 27;
    for (int dz = 0; dz < 3; ++dz) {
      int zz = z + dz - 1; if (zz < 0 || zz >= D) continue;
      for (int dy = 0; dy < 3; ++dy) {
        int yy = y + dy - 1; if (yy < 0 || yy >= D) continue;
        for (int dx = 0; dx < 3; ++dx) {
          int xx = x + dx - 1; if (xx < 0 || xx >= D) continue;
          val += ip[(zz * D + yy) * D + xx] * wp[(dz * 3 + dy) * 3 + dx];
        }
      }
    }
  }
  return val;
}

// Pass 1: conv value + per-channel sum / sumsq (BN stats). Values discarded.
__global__ void conv3d_stats(const float* __restrict__ in, const float* __restrict__ w,
                             const float* __restrict__ bias, float* __restrict__ stats,
                             int CI, int D)
{
  __shared__ float s0[256], s1[256];
  const int tid = threadIdx.x;
  const int idx = blockIdx.x * blockDim.x + tid;
  const int co  = blockIdx.y;
  const int n   = blockIdx.z;
  const int D3  = D * D * D;
  float val = 0.0f;
  bool active = idx < D3;
  if (active) {
    int z = idx / (D * D), y = (idx / D) % D, x = idx % D;
    val = conv_tap(in, w, bias[co], n, CI, D, co, z, y, x);
  }
  s0[tid] = active ? val : 0.0f;
  s1[tid] = active ? val * val : 0.0f;
  __syncthreads();
  for (int s = blockDim.x / 2; s > 0; s >>= 1) {
    if (tid < s) { s0[tid] += s0[tid + s]; s1[tid] += s1[tid + s]; }
    __syncthreads();
  }
  if (tid == 0) {
    atomicAdd(&stats[2 * co],     s0[0]);
    atomicAdd(&stats[2 * co + 1], s1[0]);
  }
}

__global__ void bn_finalize(const float* __restrict__ stats,
                            const float* __restrict__ g, const float* __restrict__ beta,
                            float* __restrict__ scale, float* __restrict__ shift,
                            int CO, float cnt)
{
  int c = blockIdx.x * blockDim.x + threadIdx.x;
  if (c >= CO) return;
  float mu  = stats[2 * c] / cnt;
  float var = stats[2 * c + 1] / cnt - mu * mu;
  float s   = g[c] * rsqrtf(var + 1e-5f);
  scale[c] = s;
  shift[c] = beta[c] - mu * s;
}

// Pass 2: recompute conv, BN-normalize, ReLU, 2x2x2 max-pool
__global__ void conv3d_bn_pool(const float* __restrict__ in, const float* __restrict__ w,
                               const float* __restrict__ bias,
                               const float* __restrict__ scale, const float* __restrict__ shift,
                               float* __restrict__ out, int CI, int D)
{
  const int Dh = D / 2, Dh3 = Dh * Dh * Dh;
  const int idx = blockIdx.x * blockDim.x + threadIdx.x;
  const int co  = blockIdx.y;
  const int CO  = gridDim.y;
  const int n   = blockIdx.z;
  if (idx >= Dh3) return;
  int z = idx / (Dh * Dh), y = (idx / Dh) % Dh, x = idx % Dh;
  float s = scale[co], t = shift[co];
  float m = 0.0f;  // relu(v) >= 0
  for (int oz = 0; oz < 2; ++oz)
    for (int oy = 0; oy < 2; ++oy)
      for (int ox = 0; ox < 2; ++ox) {
        float v = conv_tap(in, w, bias[co], n, CI, D, co,
                           2 * z + oz, 2 * y + oy, 2 * x + ox) * s + t;
        v = fmaxf(v, 0.0f);
        m = fmaxf(m, v);
      }
  out[((size_t)n * CO + co) * Dh3 + idx] = m;
}

// ---------------- small elementwise kernels ----------------
__global__ void feat_mean(const float* __restrict__ pooled, float* __restrict__ feat)
{
  int idx = blockIdx.x * blockDim.x + threadIdx.x;  // n*64+c
  if (idx >= NIMG * 64) return;
  const float* p = pooled + (size_t)idx * 8;
  float s = 0.0f;
  for (int j = 0; j < 8; ++j) s += p[j];
  feat[idx] = s * 0.125f;
}

__global__ void make_seq0(const float* __restrict__ feat, float* __restrict__ seq)
{
  int idx = blockIdx.x * blockDim.x + threadIdx.x;  // t*256*64 + e*64 + d
  if (idx >= L_PTS * E_EDGES * 64) return;
  int d = idx & 63;
  int e = (idx >> 6) & 255;
  int t = idx >> 14;
  seq[idx] = feat[((size_t)e * L_PTS + t) * 64 + d];
}

__device__ __forceinline__ float sigmoidf(float x) { return 1.0f / (1.0f + expf(-x)); }

__global__ void lstm_cell(const float* __restrict__ gates,
                          float* __restrict__ c, float* __restrict__ h,
                          float* __restrict__ hs_slot)  // hs + t*256*256 + dir*128
{
  int idx = blockIdx.x * blockDim.x + threadIdx.x;
  if (idx >= E_EDGES * LSTM_H) return;
  int e = idx >> 7, j = idx & 127;
  const float* ge = gates + (size_t)e * GATES;
  float ig = sigmoidf(ge[j]);
  float fg = sigmoidf(ge[LSTM_H + j]);
  float gg = tanhf(ge[2 * LSTM_H + j]);
  float og = sigmoidf(ge[3 * LSTM_H + j]);
  float cv = fg * c[idx] + ig * gg;
  float hv = og * tanhf(cv);
  c[idx] = cv;
  h[idx] = hv;
  hs_slot[(size_t)e * (2 * LSTM_H) + j] = hv;
}

__global__ void concat2(const float* __restrict__ a, const float* __restrict__ b,
                        float* __restrict__ out, int rows, int da, int db)
{
  int idx = blockIdx.x * blockDim.x + threadIdx.x;
  int dt = da + db;
  if (idx >= rows * dt) return;
  int r = idx / dt, j = idx % dt;
  out[idx] = (j < da) ? a[(size_t)r * da + j] : b[(size_t)r * db + (j - da)];
}

__global__ void film_ea(const float* __restrict__ g, const float* __restrict__ b,
                        const float* __restrict__ edge_attr, float* __restrict__ ea)
{
  int idx = blockIdx.x * blockDim.x + threadIdx.x;
  if (idx >= E_EDGES * EDGE_DIM) return;
  ea[idx] = g[idx] * edge_attr[idx] + b[idx];
}

__global__ void copy_f32(const float* __restrict__ a, float* __restrict__ b, int n)
{
  int idx = blockIdx.x * blockDim.x + threadIdx.x;
  if (idx < n) b[idx] = a[idx];
}

__global__ void gine_message(const float* __restrict__ xn, const float* __restrict__ eo,
                             const int* __restrict__ src, const int* __restrict__ dst,
                             float* __restrict__ aggr)
{
  int idx = blockIdx.x * blockDim.x + threadIdx.x;
  if (idx >= E_EDGES * HIDDEN) return;
  int e = idx >> 7, j = idx & 127;
  float v = xn[(size_t)src[e] * HIDDEN + j] + eo[idx];
  v = fmaxf(v, 0.0f);
  atomicAdd(&aggr[(size_t)dst[e] * HIDDEN + j], v);
}

__global__ void concat_ef(const float* __restrict__ xn, const float* __restrict__ ea,
                          const int* __restrict__ src, const int* __restrict__ dst,
                          float* __restrict__ ef)
{
  const int EF = 2 * HIDDEN + EDGE_DIM;  // 288
  int idx = blockIdx.x * blockDim.x + threadIdx.x;
  if (idx >= E_EDGES * EF) return;
  int e = idx / EF, j = idx % EF;
  float v;
  if (j < HIDDEN)            v = xn[(size_t)src[e] * HIDDEN + j];
  else if (j < 2 * HIDDEN)   v = xn[(size_t)dst[e] * HIDDEN + (j - HIDDEN)];
  else                       v = ea[(size_t)e * EDGE_DIM + (j - 2 * HIDDEN)];
  ef[idx] = v;
}

// =====================================================================
// Host side
// =====================================================================
// d_in layout per setup_inputs() insertion order (params flattened
// recursively in dict-insertion order).
enum {
  IN_IMAGES = 0, IN_X, IN_EIDX, IN_EATTR, IN_EPI,
  P_C1W, P_C1B, P_C1G, P_C1BETA,
  P_C2W, P_C2B, P_C2G, P_C2BETA,
  P_C3W, P_C3B, P_C3G, P_C3BETA,
  P_LSTM0 = 17,                // + l*8 + dir*4 + {Wih,Whh,bih,bhh}
  P_LIN1W = 49, P_LIN1B,
  P_GAMW, P_GAMB, P_BETW, P_BETB,
  P_G1 = 55,                   // + g*6 + {elin.w, elin.b, mlp1.w, mlp1.b, mlp2.w, mlp2.b}
  P_CLS1W = 73, P_CLS1B, P_CLS2W, P_CLS2B
};

static inline dim3 gemm_grid(int M, int N) {
  return dim3((N + 63) / 64, (M + 15) / 16, 1);   // 16x64 strip per wave
}

extern "C" void kernel_launch(void* const* d_in, const int* in_sizes, int n_in,
                              void* d_out, int out_size, void* d_ws, size_t ws_size,
                              hipStream_t stream)
{
  (void)in_sizes; (void)n_in; (void)out_size; (void)ws_size;
  const int BLK = 256;
  auto F = [&](int i) { return (const float*)d_in[i]; };

  const float* images = F(IN_IMAGES);
  const float* x_in   = F(IN_X);
  const int*   eidx   = (const int*)d_in[IN_EIDX];
  const float* eattr  = F(IN_EATTR);
  const int* src = eidx;
  const int* dst = eidx + E_EDGES;

  // ---- f32 workspace carve-out ----
  float* ws = (float*)d_ws;
  size_t off = 0;
  auto alloc = [&](size_t n) { float* p = ws + off; off += n; return p; };
  float* pooled1 = alloc((size_t)NIMG * 16 * 512);   // 16x8^3
  float* pooled2 = alloc((size_t)NIMG * 32 * 64);    // 32x4^3
  float* pooled3 = alloc((size_t)NIMG * 64 * 8);     // 64x2^3
  float* feat    = alloc((size_t)NIMG * 64);
  float* seq0    = alloc((size_t)L_PTS * E_EDGES * 64);
  float* hsA     = alloc((size_t)L_PTS * E_EDGES * 2 * LSTM_H);
  float* hsB     = alloc((size_t)L_PTS * E_EDGES * 2 * LSTM_H);
  float* gatesX  = alloc((size_t)L_PTS * E_EDGES * GATES);
  float* hf = alloc(E_EDGES * LSTM_H); float* cf = alloc(E_EDGES * LSTM_H);
  float* hb = alloc(E_EDGES * LSTM_H); float* cb = alloc(E_EDGES * LSTM_H);
  float* cond = alloc(E_EDGES * 2 * LSTM_H);
  float* gOut = alloc(E_EDGES * EDGE_DIM);
  float* bOut = alloc(E_EDGES * EDGE_DIM);
  float* ea   = alloc(E_EDGES * EDGE_DIM);
  float* xn   = alloc(N_NODES * HIDDEN);
  float* aggr = alloc(N_NODES * HIDDEN);
  float* t1   = alloc(N_NODES * HIDDEN);
  float* xn2  = alloc(N_NODES * HIDDEN);
  float* eo   = alloc(E_EDGES * HIDDEN);
  float* ef   = alloc((size_t)E_EDGES * (2 * HIDDEN + EDGE_DIM));
  float* t2   = alloc(E_EDGES * HIDDEN);
  float* stats1 = alloc(32);  float* stats2 = alloc(64);  float* stats3 = alloc(128);
  float* sc1 = alloc(16); float* sh1 = alloc(16);
  float* sc2 = alloc(32); float* sh2 = alloc(32);
  float* sc3 = alloc(64); float* sh3 = alloc(64);

  // ---- bf16 weight workspace (32B-aligned: off is a multiple of 8) ----
  off = (off + 7) & ~(size_t)7;
  __bf16* wsbf = (__bf16*)(ws + off);
  size_t boff = 0;
  auto balloc = [&](size_t n) { __bf16* p = wsbf + boff; boff += n; return p; };
  __bf16* wih_bf[4][2]; __bf16* whh_bf[4][2];
  for (int l = 0; l < 4; ++l)
    for (int d = 0; d < 2; ++d) {
      wih_bf[l][d] = balloc((size_t)GATES * (l == 0 ? 64 : 256));
      whh_bf[l][d] = balloc((size_t)GATES * LSTM_H);
    }
  __bf16* lin1_bf = balloc((size_t)HIDDEN * NODE_DIM);
  __bf16* gam_bf  = balloc((size_t)EDGE_DIM * 2 * LSTM_H);
  __bf16* bet_bf  = balloc((size_t)EDGE_DIM * 2 * LSTM_H);
  __bf16* elin_bf[3]; __bf16* mlp1_bf[3]; __bf16* mlp2_bf[3];
  for (int g = 0; g < 3; ++g) {
    elin_bf[g] = balloc((size_t)HIDDEN * EDGE_DIM);
    mlp1_bf[g] = balloc((size_t)HIDDEN * HIDDEN);
    mlp2_bf[g] = balloc((size_t)HIDDEN * HIDDEN);
  }
  __bf16* cls1_bf = balloc((size_t)HIDDEN * (2 * HIDDEN + EDGE_DIM));
  __bf16* cls2_bf = balloc((size_t)OUT_DIM * LSTM_H);

  auto cvt = [&](const float* srcp, __bf16* dstp, int n) {
    f32_to_bf16<<<(n + BLK - 1) / BLK, BLK, 0, stream>>>(srcp, dstp, n);
  };
  for (int l = 0; l < 4; ++l)
    for (int d = 0; d < 2; ++d) {
      cvt(F(P_LSTM0 + l * 8 + d * 4 + 0), wih_bf[l][d], GATES * (l == 0 ? 64 : 256));
      cvt(F(P_LSTM0 + l * 8 + d * 4 + 1), whh_bf[l][d], GATES * LSTM_H);
    }
  cvt(F(P_LIN1W), lin1_bf, HIDDEN * NODE_DIM);
  cvt(F(P_GAMW), gam_bf, EDGE_DIM * 2 * LSTM_H);
  cvt(F(P_BETW), bet_bf, EDGE_DIM * 2 * LSTM_H);
  for (int g = 0; g < 3; ++g) {
    cvt(F(P_G1 + g * 6 + 0), elin_bf[g], HIDDEN * EDGE_DIM);
    cvt(F(P_G1 + g * 6 + 2), mlp1_bf[g], HIDDEN * HIDDEN);
    cvt(F(P_G1 + g * 6 + 4), mlp2_bf[g], HIDDEN * HIDDEN);
  }
  cvt(F(P_CLS1W), cls1_bf, HIDDEN * (2 * HIDDEN + EDGE_DIM));
  cvt(F(P_CLS2W), cls2_bf, OUT_DIM * LSTM_H);

  hipMemsetAsync(stats1, 0, 32 * sizeof(float), stream);
  hipMemsetAsync(stats2, 0, 64 * sizeof(float), stream);
  hipMemsetAsync(stats3, 0, 128 * sizeof(float), stream);

  // ---- CNN tower: conv -> BN stats -> (recompute) BN + ReLU + pool ----
  conv3d_stats<<<dim3(16, 16, NIMG), BLK, 0, stream>>>(images, F(P_C1W), F(P_C1B), stats1, 1, 16);
  bn_finalize<<<1, 16, 0, stream>>>(stats1, F(P_C1G), F(P_C1BETA), sc1, sh1, 16, (float)NIMG * 4096.0f);
  conv3d_bn_pool<<<dim3(2, 16, NIMG), BLK, 0, stream>>>(images, F(P_C1W), F(P_C1B), sc1, sh1, pooled1, 1, 16);
  conv3d_stats<<<dim3(2, 32, NIMG), BLK, 0, stream>>>(pooled1, F(P_C2W), F(P_C2B), stats2, 16, 8);
  bn_finalize<<<1, 32, 0, stream>>>(stats2, F(P_C2G), F(P_C2BETA), sc2, sh2, 32, (float)NIMG * 512.0f);
  conv3d_bn_pool<<<dim3(1, 32, NIMG), BLK, 0, stream>>>(pooled1, F(P_C2W), F(P_C2B), sc2, sh2, pooled2, 16, 8);
  conv3d_stats<<<dim3(1, 64, NIMG), BLK, 0, stream>>>(pooled2, F(P_C3W), F(P_C3B), stats3, 32, 4);
  bn_finalize<<<1, 64, 0, stream>>>(stats3, F(P_C3G), F(P_C3BETA), sc3, sh3, 64, (float)NIMG * 64.0f);
  conv3d_bn_pool<<<dim3(1, 64, NIMG), BLK, 0, stream>>>(pooled2, F(P_C3W), F(P_C3B), sc3, sh3, pooled3, 32, 4);

  feat_mean<<<(NIMG * 64 + BLK - 1) / BLK, BLK, 0, stream>>>(pooled3, feat);
  make_seq0<<<(L_PTS * E_EDGES * 64 + BLK - 1) / BLK, BLK, 0, stream>>>(feat, seq0);

  // ---- 4-layer bidirectional LSTM (WMMA GEMMs) ----
  float* hs_prev = nullptr;
  for (int l = 0; l < 4; ++l) {
    int in_dim = (l == 0) ? 64 : 2 * LSTM_H;
    const float* Xl = (l == 0) ? seq0 : hs_prev;
    float* hs_cur = (l & 1) ? hsB : hsA;
    for (int dir = 0; dir < 2; ++dir) {
      const float* bih = F(P_LSTM0 + l * 8 + dir * 4 + 2);
      const float* bhh = F(P_LSTM0 + l * 8 + dir * 4 + 3);
      float* h = dir ? hb : hf;
      float* c = dir ? cb : cf;
      // gatesX[8*256, 512] = X @ Wih^T + bih + bhh
      wmma_gemm_bf16<<<gemm_grid(L_PTS * E_EDGES, GATES), 32, 0, stream>>>(
          Xl, wih_bf[l][dir], bih, bhh, gatesX, L_PTS * E_EDGES, GATES, in_dim, 0, 0);
      hipMemsetAsync(h, 0, E_EDGES * LSTM_H * sizeof(float), stream);
      hipMemsetAsync(c, 0, E_EDGES * LSTM_H * sizeof(float), stream);
      for (int s = 0; s < 8; ++s) {
        int t = dir ? (7 - s) : s;
        float* gt = gatesX + (size_t)t * E_EDGES * GATES;
        // gt += h @ Whh^T   (accumulate into precomputed X-gates)
        wmma_gemm_bf16<<<gemm_grid(E_EDGES, GATES), 32, 0, stream>>>(
            h, whh_bf[l][dir], nullptr, nullptr, gt, E_EDGES, GATES, LSTM_H, 1, 0);
        lstm_cell<<<(E_EDGES * LSTM_H + BLK - 1) / BLK, BLK, 0, stream>>>(
            gt, c, h, hs_cur + (size_t)t * E_EDGES * 2 * LSTM_H + dir * LSTM_H);
      }
    }
    hs_prev = hs_cur;
  }

  // cond = concat(hTf, hTb)  (256 x 256)
  concat2<<<(E_EDGES * 2 * LSTM_H + BLK - 1) / BLK, BLK, 0, stream>>>(
      hf, hb, cond, E_EDGES, LSTM_H, LSTM_H);

  // ea = lin(cond, gamma) * edge_attr + lin(cond, beta)
  wmma_gemm_bf16<<<gemm_grid(E_EDGES, EDGE_DIM), 32, 0, stream>>>(
      cond, gam_bf, F(P_GAMB), nullptr, gOut, E_EDGES, EDGE_DIM, 2 * LSTM_H, 0, 0);
  wmma_gemm_bf16<<<gemm_grid(E_EDGES, EDGE_DIM), 32, 0, stream>>>(
      cond, bet_bf, F(P_BETB), nullptr, bOut, E_EDGES, EDGE_DIM, 2 * LSTM_H, 0, 0);
  film_ea<<<(E_EDGES * EDGE_DIM + BLK - 1) / BLK, BLK, 0, stream>>>(gOut, bOut, eattr, ea);

  // xn = x @ lin1^T + b   (300 x 128)
  wmma_gemm_bf16<<<gemm_grid(N_NODES, HIDDEN), 32, 0, stream>>>(
      x_in, lin1_bf, F(P_LIN1B), nullptr, xn, N_NODES, HIDDEN, NODE_DIM, 0, 0);

  // ---- 3x GINE layers ----
  for (int g = 0; g < 3; ++g) {
    const float* elb = F(P_G1 + g * 6 + 1);
    const float* m1b = F(P_G1 + g * 6 + 3);
    const float* m2b = F(P_G1 + g * 6 + 5);
    wmma_gemm_bf16<<<gemm_grid(E_EDGES, HIDDEN), 32, 0, stream>>>(
        ea, elin_bf[g], elb, nullptr, eo, E_EDGES, HIDDEN, EDGE_DIM, 0, 0);
    copy_f32<<<(N_NODES * HIDDEN + BLK - 1) / BLK, BLK, 0, stream>>>(xn, aggr, N_NODES * HIDDEN);
    gine_message<<<(E_EDGES * HIDDEN + BLK - 1) / BLK, BLK, 0, stream>>>(xn, eo, src, dst, aggr);
    wmma_gemm_bf16<<<gemm_grid(N_NODES, HIDDEN), 32, 0, stream>>>(
        aggr, mlp1_bf[g], m1b, nullptr, t1, N_NODES, HIDDEN, HIDDEN, 0, 1);
    wmma_gemm_bf16<<<gemm_grid(N_NODES, HIDDEN), 32, 0, stream>>>(
        t1, mlp2_bf[g], m2b, nullptr, xn2, N_NODES, HIDDEN, HIDDEN, 0, 1);
    float* tmp = xn; xn = xn2; xn2 = tmp;
  }

  // ---- classifier ----
  const int EF = 2 * HIDDEN + EDGE_DIM;  // 288
  concat_ef<<<(E_EDGES * EF + BLK - 1) / BLK, BLK, 0, stream>>>(xn, ea, src, dst, ef);
  wmma_gemm_bf16<<<gemm_grid(E_EDGES, HIDDEN), 32, 0, stream>>>(
      ef, cls1_bf, F(P_CLS1B), nullptr, t2, E_EDGES, HIDDEN, EF, 0, 1);
  wmma_gemm_bf16<<<gemm_grid(E_EDGES, OUT_DIM), 32, 0, stream>>>(
      t2, cls2_bf, F(P_CLS2B), nullptr, (float*)d_out, E_EDGES, OUT_DIM, LSTM_H, 0, 0);
}
